// BiLSTM_CRF_65257733095633
// MI455X (gfx1250) — matrix operand order repaired and checked by reading, not compiled, not run on hardware
//
#include <hip/hip_runtime.h>
#include <hip/hip_bf16.h>
#include <cstddef>

typedef __attribute__((ext_vector_type(16))) _Float16 v16h;
typedef __attribute__((ext_vector_type(8)))  _Float16 v8h;
typedef __attribute__((ext_vector_type(8)))  float    v8f;

#define TT 256
#define BB 64
#define EE 256
#define HH 512
#define G4 2048
#define NN 6
#define TB (TT*BB)        // 16384
#define NB_REC 16         // blocks per direction in recurrence
#define WAVES_PER_BLOCK 8

// ---------------- WMMA helpers (CDNA5 16x16x32 f16 -> f32) ----------------

__device__ __forceinline__ v8f wmma16x16x32(v16h a, v16h b, v8f c) {
    return __builtin_amdgcn_wmma_f32_16x16x32_f16(false, a, false, b, (short)0, c, false, false);
}

// A fragment: 16(M) x 32(K) f16, row-major source with leading dim ld.
// lanes 0-15: M=lane, K = {k0..k0+7, k0+16..k0+23}
// lanes 16-31: M=lane-16, K = {k0+8..k0+15, k0+24..k0+31}
__device__ __forceinline__ v16h load_a_frag(const _Float16* __restrict__ base, int ld,
                                            int mbase, int k0, int lane) {
    const int row = mbase + (lane & 15);
    const int kb  = k0 + ((lane & 16) ? 8 : 0);
    const _Float16* p = base + (size_t)row * ld + kb;
    v8h lo = *(const v8h*)(p);
    v8h hi = *(const v8h*)(p + 16);
    return __builtin_shufflevector(lo, hi, 0,1,2,3,4,5,6,7,8,9,10,11,12,13,14,15);
}

// B fragment: 32(K) x 16(N); output col n = row of W (G = A*W^T), so each lane
// loads 16 K-values (two runs of 8) of one W row.
// lanes 0-15: N=lane, K = {k0..k0+7, k0+16..k0+23} ; lanes 16-31: N=lane-16, K = {k0+8..k0+15, k0+24..k0+31}
__device__ __forceinline__ v16h load_b_frag(const _Float16* __restrict__ base, int ld,
                                            int colbase, int k0, int lane) {
    const int col = colbase + (lane & 15);
    const int kb  = k0 + ((lane & 16) ? 8 : 0);
    const _Float16* p = base + (size_t)col * ld + kb;
    v8h lo = *(const v8h*)(p);
    v8h hi = *(const v8h*)(p + 16);
    return __builtin_shufflevector(lo, hi, 0,1,2,3,4,5,6,7,8,9,10,11,12,13,14,15);
}

__device__ __forceinline__ float sigf(float x) { return 1.0f / (1.0f + __expf(-x)); }

// ---------------- prep kernels ----------------

__global__ void f32_to_f16_kernel(const float* __restrict__ src, _Float16* __restrict__ dst, int n) {
    int i = blockIdx.x * blockDim.x + threadIdx.x;
    if (i < n) dst[i] = (_Float16)src[i];
}

__global__ void gather_embed_kernel(const int* __restrict__ sent,
                                    const float* __restrict__ embed,
                                    _Float16* __restrict__ xh) {
    int tid = blockIdx.x * blockDim.x + threadIdx.x;   // TB*64 threads, 4 elems each
    int tok = tid >> 6;
    int e4  = (tid & 63) * 4;
    int w = sent[tok];
    float4 v = *(const float4*)(embed + (size_t)w * EE + e4);
    _Float16* p = xh + (size_t)tok * EE + e4;
    p[0] = (_Float16)v.x; p[1] = (_Float16)v.y;
    p[2] = (_Float16)v.z; p[3] = (_Float16)v.w;
}

__global__ void init_state_kernel(const float* __restrict__ h0, const float* __restrict__ c0,
                                  _Float16* __restrict__ hbuf, float* __restrict__ cbuf,
                                  int* __restrict__ bar) {
    int i = blockIdx.x * blockDim.x + threadIdx.x;     // 2*BB*HH = 65536 threads
    if (i == 0) { bar[0] = 0; bar[16] = 0; }
    int dir = i >> 15;
    int off = i & 32767;                               // b*HH + hcol
    hbuf[(size_t)dir * 2 * (BB*HH) + off] = (_Float16)h0[(size_t)dir * (BB*HH) + off]; // buffer 0
    cbuf[(size_t)dir * (BB*HH) + off]     = c0[(size_t)dir * (BB*HH) + off];
}

// ---------------- xg = X * W_ih^T + b : M=TB, K=EE, N=G4, per direction --------------
// Each wave owns 2 M-tiles x 4 N-tiles (8 accumulators): per k-iter it loads
// 2 A + 4 B fragments and issues 8 independent WMMAs, so load waits overlap with
// matrix work instead of serializing. Output stored in WMMA C-fragment layout:
// xg[((mtile*128 + ntile)*32 + lane)*8 + r], mtile = t*4+btile, ntile = gate*32+hct.

__global__ void xg_gemm_kernel(const _Float16* __restrict__ xh,
                               const _Float16* __restrict__ WihF, const _Float16* __restrict__ WihB,
                               const float* __restrict__ bF, const float* __restrict__ bB,
                               float* __restrict__ xgF, float* __restrict__ xgB) {
    const int dir = blockIdx.y;
    const _Float16* W = dir ? WihB : WihF;
    const float* bias = dir ? bB : bF;
    float* out = dir ? xgB : xgF;

    const int lane = threadIdx.x & 31;
    const int wave = threadIdx.x >> 5;
    const int job  = blockIdx.x * WAVES_PER_BLOCK + wave;  // 0..16383
    const int mp   = job >> 5;                             // 512 m-tile pairs
    const int ng   = job & 31;                             // 32 n-groups of 4 tiles
    const int mbase = mp * 32;
    const int nbase = ng * 64;

    v8f acc[2][4];
    #pragma unroll
    for (int mi = 0; mi < 2; ++mi)
        #pragma unroll
        for (int s = 0; s < 4; ++s)
            acc[mi][s] = (v8f){0.f,0.f,0.f,0.f,0.f,0.f,0.f,0.f};

    #pragma unroll
    for (int k0 = 0; k0 < EE; k0 += 32) {
        v16h a0 = load_a_frag(xh, EE, mbase,      k0, lane);
        v16h a1 = load_a_frag(xh, EE, mbase + 16, k0, lane);
        v16h b0 = load_b_frag(W, EE, nbase +  0, k0, lane);
        v16h b1 = load_b_frag(W, EE, nbase + 16, k0, lane);
        v16h b2 = load_b_frag(W, EE, nbase + 32, k0, lane);
        v16h b3 = load_b_frag(W, EE, nbase + 48, k0, lane);
        acc[0][0] = wmma16x16x32(a0, b0, acc[0][0]);
        acc[1][0] = wmma16x16x32(a1, b0, acc[1][0]);
        acc[0][1] = wmma16x16x32(a0, b1, acc[0][1]);
        acc[1][1] = wmma16x16x32(a1, b1, acc[1][1]);
        acc[0][2] = wmma16x16x32(a0, b2, acc[0][2]);
        acc[1][2] = wmma16x16x32(a1, b2, acc[1][2]);
        acc[0][3] = wmma16x16x32(a0, b3, acc[0][3]);
        acc[1][3] = wmma16x16x32(a1, b3, acc[1][3]);
    }

    const int n = lane & 15;
    #pragma unroll
    for (int mi = 0; mi < 2; ++mi) {
        const size_t mtile = (size_t)mp * 2 + mi;
        #pragma unroll
        for (int s = 0; s < 4; ++s) {
            const int ntile = ng * 4 + s;
            const float bs = bias[ntile * 16 + n];
            v8f res;
            #pragma unroll
            for (int r = 0; r < 8; ++r) res[r] = acc[mi][s][r] + bs;
            *(v8f*)(out + ((mtile * 128 + ntile) * 32 + lane) * 8) = res;
        }
    }
}

// ---------------- persistent bidirectional LSTM recurrence ----------------
// grid = (NB_REC, 2). Each block owns 2 h-column tiles (32 cols) for ALL batch tiles
// and gates. Its W_hh slice (4 gates x 32 cols x 512 K fp16 = 128KB) is staged in LDS
// ONCE for all 256 steps; h (64KB fp16) is staged in LDS once per step. All WMMA
// operands come from LDS on the serial critical path; the 16-iteration k-loop is
// fully unrolled so ds_load offsets are immediates. xg arrives as one contiguous
// 32B fragment load per gate per lane, issued before the k-loop to hide latency.

__global__ void lstm_rec_kernel(const float* __restrict__ xgF, const float* __restrict__ xgB,
                                const _Float16* __restrict__ WhhF, const _Float16* __restrict__ WhhB,
                                const int* __restrict__ mask,
                                _Float16* __restrict__ hbuf,   // [2 dir][2 buf][BB*HH]
                                float* __restrict__ cbuf,      // [2 dir][BB*HH]
                                float* __restrict__ hsF, float* __restrict__ hsB,
                                int* __restrict__ bar) {
    extern __shared__ _Float16 smem[];
    _Float16* Wlds = smem;                      // [4*32][HH] = 128KB
    _Float16* hlds = smem + 4 * 32 * HH;        // [BB][HH]   =  64KB

    const int dir = blockIdx.y;
    const float* xg = dir ? xgB : xgF;
    const _Float16* Whh = dir ? WhhB : WhhF;
    float* hs = dir ? hsB : hsF;
    _Float16* hb = hbuf + (size_t)dir * 2 * (BB*HH);
    float* c = cbuf + (size_t)dir * (BB*HH);
    int* mybar = bar + dir * 16;

    const int bx   = blockIdx.x;                // 0..15
    const int lane = threadIdx.x & 31;
    const int wave = threadIdx.x >> 5;          // 0..7
    const int mt   = wave & 3;                  // batch tile
    const int lc   = wave >> 2;                 // local hcol tile 0..1
    const int hct  = bx * 2 + lc;               // global hcol tile 0..31
    const int mbase = mt * 16;
    const int hcb_l = lc * 16;

    // Stage this block's W_hh slice into LDS (once, reused for all 256 steps).
    {
        const float4* src = (const float4*)Whh;
        float4* dst = (float4*)Wlds;
        #pragma unroll
        for (int j = 0; j < 32; ++j) {
            const int fi = threadIdx.x + j * 256;      // 0..8191 float4s
            const int rl = fi >> 6;                    // local row 0..127 (g*32+cl)
            const int k8 = fi & 63;
            const int rg = (rl >> 5) * HH + bx * 32 + (rl & 31);   // W_hh row
            dst[fi] = src[(size_t)rg * 64 + k8];
        }
    }
    __syncthreads();

    for (int step = 0; step < TT; ++step) {
        const int t = dir ? (TT - 1 - step) : step;
        const _Float16* hcur = hb + (size_t)(step & 1) * (BB*HH);
        _Float16* hnxt = hb + (size_t)((step + 1) & 1) * (BB*HH);

        // Stage h into LDS (64KB: 16 float4 per thread).
        {
            const float4* src = (const float4*)hcur;
            float4* dst = (float4*)hlds;
            #pragma unroll
            for (int j = 0; j < 16; ++j)
                dst[threadIdx.x + j * 256] = src[threadIdx.x + j * 256];
        }
        __syncthreads();

        // Prefetch xg fragments (global, contiguous 32B per gate) before the GEMM.
        v8f xgv[4];
        const size_t tilebase = ((size_t)t * 4 + mt) * 128;
        #pragma unroll
        for (int g = 0; g < 4; ++g)
            xgv[g] = *(const v8f*)(xg + ((tilebase + g * 32 + hct) * 32 + lane) * 8);

        v8f acc[4];
        #pragma unroll
        for (int g = 0; g < 4; ++g) acc[g] = (v8f){0.f,0.f,0.f,0.f,0.f,0.f,0.f,0.f};

        #pragma unroll
        for (int k0 = 0; k0 < HH; k0 += 32) {
            v16h a  = load_a_frag(hlds, HH, mbase, k0, lane);
            v16h w0 = load_b_frag(Wlds, HH, 0 * 32 + hcb_l, k0, lane);
            v16h w1 = load_b_frag(Wlds, HH, 1 * 32 + hcb_l, k0, lane);
            v16h w2 = load_b_frag(Wlds, HH, 2 * 32 + hcb_l, k0, lane);
            v16h w3 = load_b_frag(Wlds, HH, 3 * 32 + hcb_l, k0, lane);
            acc[0] = wmma16x16x32(a, w0, acc[0]);
            acc[1] = wmma16x16x32(a, w1, acc[1]);
            acc[2] = wmma16x16x32(a, w2, acc[2]);
            acc[3] = wmma16x16x32(a, w3, acc[3]);
        }

        // LSTM cell update: i/f/g/o for each (b,hcol) live in the same lane.
        const int n  = lane & 15;
        const int m0 = (lane & 16) ? 8 : 0;
        const int hcol = hct * 16 + n;
        #pragma unroll
        for (int r = 0; r < 8; ++r) {
            const int b = mbase + m0 + r;
            float iv = acc[0][r] + xgv[0][r];
            float fv = acc[1][r] + xgv[1][r];
            float gv = acc[2][r] + xgv[2][r];
            float ov = acc[3][r] + xgv[3][r];
            float cold = c[b * HH + hcol];
            float cnew = sigf(fv) * cold + sigf(iv) * tanhf(gv);
            float hnew = sigf(ov) * tanhf(cnew);
            float mm = (float)mask[t * BB + b];
            float hold = (float)hlds[b * HH + hcol];
            float h2 = mm * hnew + (1.0f - mm) * hold;
            float c2 = mm * cnew + (1.0f - mm) * cold;
            c[b * HH + hcol] = c2;
            hnxt[b * HH + hcol] = (_Float16)h2;
            hs[((size_t)t * BB + b) * HH + hcol] = h2 * mm;
        }

        // device-scope barrier among the NB_REC blocks of this direction
        __threadfence();
        __syncthreads();
        if (threadIdx.x == 0) {
            atomicAdd(mybar, 1);
            const int target = NB_REC * (step + 1);
            while (atomicAdd(mybar, 0) < target) { }
        }
        __syncthreads();
        __threadfence();
    }
}

// ---------------- emissions: (hf|hb) @ W_out^T + b_out, N=6 ----------------

__global__ void emis_kernel(const float* __restrict__ hsF, const float* __restrict__ hsB,
                            const float* __restrict__ Wout, const float* __restrict__ bout,
                            float* __restrict__ emis) {
    int tid = blockIdx.x * blockDim.x + threadIdx.x;
    if (tid >= TB * NN) return;
    int n  = tid % NN;
    int tb = tid / NN;
    const float* hf  = hsF + (size_t)tb * HH;
    const float* hbp = hsB + (size_t)tb * HH;
    const float* w0 = Wout + (size_t)n * (2 * HH);
    const float* w1 = w0 + HH;
    float s = bout[n];
    for (int k = 0; k < HH; k += 4) {
        float4 a = *(const float4*)(hf + k);
        float4 w = *(const float4*)(w0 + k);
        s += a.x * w.x + a.y * w.y + a.z * w.z + a.w * w.w;
    }
    for (int k = 0; k < HH; k += 4) {
        float4 a = *(const float4*)(hbp + k);
        float4 w = *(const float4*)(w1 + k);
        s += a.x * w.x + a.y * w.y + a.z * w.z + a.w * w.w;
    }
    emis[tid] = s;
}

// ---------------- CRF log-likelihood: one lane per batch element ----------------

__global__ void crf_kernel(const float* __restrict__ emis, const int* __restrict__ tags,
                           const int* __restrict__ mask, const float* __restrict__ trans,
                           const float* __restrict__ startt, const float* __restrict__ endt,
                           float* __restrict__ out) {
    __shared__ float str[NN * NN], ss[NN], se[NN], red[BB];
    const int tid = threadIdx.x;
    if (tid < NN * NN) str[tid] = trans[tid];
    if (tid < NN) { ss[tid] = startt[tid]; se[tid] = endt[tid]; }
    __syncthreads();

    const int b = tid;
    float s[NN];
    const int tg0 = tags[b];
    float num = ss[tg0] + emis[(size_t)b * NN + tg0];
    #pragma unroll
    for (int j = 0; j < NN; ++j) s[j] = ss[j] + emis[(size_t)b * NN + j];
    int prev = tg0;
    int cnt = (mask[b] != 0) ? 1 : 0;

    for (int t = 1; t < TT; ++t) {
        const float* e = emis + ((size_t)t * BB + b) * NN;
        const int cur = tags[t * BB + b];
        const int mt  = mask[t * BB + b];
        const float mf = (float)mt;
        num += mf * (str[prev * NN + cur] + e[cur]);
        prev = cur;
        cnt += (mt != 0);

        float nxt[NN];
        #pragma unroll
        for (int j = 0; j < NN; ++j) {
            float mx = -3.0e38f;
            #pragma unroll
            for (int i = 0; i < NN; ++i) mx = fmaxf(mx, s[i] + str[i * NN + j]);
            float sm = 0.0f;
            #pragma unroll
            for (int i = 0; i < NN; ++i) sm += __expf(s[i] + str[i * NN + j] - mx);
            nxt[j] = mx + __logf(sm) + e[j];
        }
        if (mt > 0) {
            #pragma unroll
            for (int j = 0; j < NN; ++j) s[j] = nxt[j];
        }
    }

    const int last = cnt - 1;
    num += se[tags[last * BB + b]];
    float mx = -3.0e38f;
    #pragma unroll
    for (int j = 0; j < NN; ++j) mx = fmaxf(mx, s[j] + se[j]);
    float sm = 0.0f;
    #pragma unroll
    for (int j = 0; j < NN; ++j) sm += __expf(s[j] + se[j] - mx);
    red[b] = num - (mx + __logf(sm));
    __syncthreads();
    if (tid == 0) {
        float tot = 0.0f;
        for (int i = 0; i < BB; ++i) tot += red[i];
        out[0] = tot;
    }
}

// ---------------- host ----------------

extern "C" void kernel_launch(void* const* d_in, const int* in_sizes, int n_in,
                              void* d_out, int out_size, void* d_ws, size_t ws_size,
                              hipStream_t stream) {
    const int*   sentence   = (const int*)d_in[0];
    const int*   tags       = (const int*)d_in[1];
    const int*   mask       = (const int*)d_in[2];
    const float* h0         = (const float*)d_in[3];
    const float* c0         = (const float*)d_in[4];
    const float* embed      = (const float*)d_in[5];
    const float* W_ih_f     = (const float*)d_in[6];
    const float* W_hh_f     = (const float*)d_in[7];
    const float* b_f        = (const float*)d_in[8];
    const float* W_ih_b     = (const float*)d_in[9];
    const float* W_hh_b     = (const float*)d_in[10];
    const float* b_b        = (const float*)d_in[11];
    const float* W_out      = (const float*)d_in[12];
    const float* b_out      = (const float*)d_in[13];
    const float* trans      = (const float*)d_in[14];
    const float* start_tr   = (const float*)d_in[15];
    const float* end_tr     = (const float*)d_in[16];
    float* out = (float*)d_out;

    char* ws = (char*)d_ws;
    size_t off = 0;
    auto alloc = [&](size_t bytes) -> void* {
        void* p = (void*)(ws + off);
        off = (off + bytes + 255) & ~(size_t)255;
        return p;
    };

    _Float16* xh    = (_Float16*)alloc((size_t)TB * EE * 2);
    _Float16* WihFh = (_Float16*)alloc((size_t)G4 * EE * 2);
    _Float16* WihBh = (_Float16*)alloc((size_t)G4 * EE * 2);
    _Float16* WhhFh = (_Float16*)alloc((size_t)G4 * HH * 2);
    _Float16* WhhBh = (_Float16*)alloc((size_t)G4 * HH * 2);
    float*    xgF   = (float*)   alloc((size_t)TB * G4 * 4);
    float*    xgB   = (float*)   alloc((size_t)TB * G4 * 4);
    _Float16* hbuf  = (_Float16*)alloc((size_t)2 * 2 * BB * HH * 2);
    float*    cbuf  = (float*)   alloc((size_t)2 * BB * HH * 4);
    float*    hsF   = (float*)   alloc((size_t)TB * HH * 4);
    float*    hsB   = (float*)   alloc((size_t)TB * HH * 4);
    float*    emis  = (float*)   alloc((size_t)TB * NN * 4);
    int*      bar   = (int*)     alloc(256);

    // 1. weight conversions to fp16
    f32_to_f16_kernel<<<(G4*EE + 255)/256, 256, 0, stream>>>(W_ih_f, WihFh, G4*EE);
    f32_to_f16_kernel<<<(G4*EE + 255)/256, 256, 0, stream>>>(W_ih_b, WihBh, G4*EE);
    f32_to_f16_kernel<<<(G4*HH + 255)/256, 256, 0, stream>>>(W_hh_f, WhhFh, G4*HH);
    f32_to_f16_kernel<<<(G4*HH + 255)/256, 256, 0, stream>>>(W_hh_b, WhhBh, G4*HH);

    // 2. embedding gather + fp16 convert
    gather_embed_kernel<<<(TB * 64) / 256, 256, 0, stream>>>(sentence, embed, xh);

    // 3. init h/c state + barrier counters
    init_state_kernel<<<(2 * BB * HH) / 256, 256, 0, stream>>>(h0, c0, hbuf, cbuf, bar);

    // 4. input-projection GEMM (both directions): 2x4 tiles per wave
    {
        dim3 grid((TB/32) * (G4/64) / WAVES_PER_BLOCK, 2);   // (2048, 2)
        xg_gemm_kernel<<<grid, 256, 0, stream>>>(xh, WihFh, WihBh, b_f, b_b, xgF, xgB);
    }

    // 5. persistent bidirectional recurrence (128KB W slice + 64KB h in LDS)
    {
        const size_t lds_bytes = (size_t)(4 * 32 * HH + BB * HH) * sizeof(_Float16); // 192KB
        hipFuncSetAttribute((const void*)lstm_rec_kernel,
                            hipFuncAttributeMaxDynamicSharedMemorySize,
                            (int)lds_bytes);
        dim3 grid(NB_REC, 2);
        lstm_rec_kernel<<<grid, 256, lds_bytes, stream>>>(xgF, xgB, WhhFh, WhhBh, mask,
                                                          hbuf, cbuf, hsF, hsB, bar);
    }

    // 6. emissions
    emis_kernel<<<(TB * NN + 255)/256, 256, 0, stream>>>(hsF, hsB, W_out, b_out, emis);

    // 7. CRF log-likelihood
    crf_kernel<<<1, BB, 0, stream>>>(emis, tags, mask, trans, start_tr, end_tr, out);
}
